// Embedder_3238405341783
// MI455X (gfx1250) — compile-verified, matrix operand order checked by reading, not run on hardware
//
#include <hip/hip_runtime.h>
#include <hip/hip_bf16.h>
#include <math.h>

// ---------------------------------------------------------------------------
// CDNA5 (gfx1250) graph-transformer forward pass:
//  - all GEMMs on v_wmma_f32_16x16x32_f16 (f16 in, f32 accumulate)
//  - TDM (tensor_load_to_lds) stages the weight tile into LDS, with TDM
//    padding generating the bank-conflict-avoiding row stride
//  - flash attention with dh=32 == one WMMA K-step
// ---------------------------------------------------------------------------

typedef _Float16 half16 __attribute__((ext_vector_type(16)));
typedef float    float8 __attribute__((ext_vector_type(8)));
typedef unsigned int uint32x4 __attribute__((ext_vector_type(4)));
typedef int          int32x4 __attribute__((ext_vector_type(4)));
typedef int          int32x8 __attribute__((ext_vector_type(8)));

#define N_NODES  16384
#define N_EDGES  262144
#define N_GRAPHS 64
#define COLORS   16
#define EMB      256
#define HEADS    8
#define DH       32
#define FFN_DIM  1024
#define LAYERS   4
#define LMAX     320
#define LG       (LMAX * N_GRAPHS)   // 20480 packed rows

#if defined(__has_builtin)
#if __has_builtin(__builtin_amdgcn_tensor_load_to_lds) && \
    __has_builtin(__builtin_amdgcn_s_wait_tensorcnt)
#define HAS_TDM 1
#endif
#endif
#ifndef HAS_TDM
#define HAS_TDM 0
#endif

// K-index inside a 16x32 f16 WMMA A/B fragment (wave32 layout, ISA 7.12.2):
// lane group g = lane/16 ; element e in 0..15 ->
//   k = 8*g + ((e>>3)<<4) + (e&7)
__device__ __forceinline__ int wk_idx(int e, int g) {
  return 8 * g + ((e >> 3) << 4) + (e & 7);
}

// ---------------------------------------------------------------------------
// GEMM:  C[M,N] = act( A_f32[M,K] @ B_f16[K,N] (+bias) (+Cadd) )
// 256 threads = 8 waves; block tile 128x64; wave grid 4x2, each wave owns a
// 32x32 output (2x2 accumulators -> 4 back-to-back WMMAs per K-step).
// B tile (32 rows x 64 f16) staged by the Tensor Data Mover when K is a
// multiple of 32; TDM pad feature (pad_interval=32 DW, pad_amount=4 DW)
// produces the padded 72-half LDS row stride.
// ---------------------------------------------------------------------------
template <bool RELU, bool ADDC, bool BIAS, typename OutT>
__launch_bounds__(256)
__global__ void gemm_wmma_kernel(const float* __restrict__ A,
                                 const _Float16* __restrict__ B,
                                 const float* __restrict__ bias,
                                 const float* __restrict__ Cadd,
                                 OutT* __restrict__ C,
                                 int M, int N, int K) {
  __shared__ _Float16 As[128][40];   // padded to dodge bank conflicts
  __shared__ _Float16 Bs[32][72];

  const int tid  = threadIdx.x;
  const int lane = tid & 31;
  const int wave = tid >> 5;        // 0..7
  const int wm   = wave >> 1;       // 0..3  (32-row slab within block)
  const int wn   = wave & 1;        // 0..1  (32-col slab within block)
  const int g    = lane >> 4;
  const int mr   = lane & 15;
  const int blockM = blockIdx.x * 128;
  const int blockN = blockIdx.y * 64;
  const bool fullK = (K & 31) == 0;
  (void)fullK;

  float8 acc[2][2] = {};

  for (int k0 = 0; k0 < K; k0 += 32) {
    // prefetch next A tile into cache while this step computes
    if (k0 + 32 < K) {
      const float* nxt =
          A + (size_t)(blockM + (tid >> 1)) * K + (k0 + 32) + (tid & 1) * 16;
      __builtin_prefetch(nxt, 0, 0);   // -> global_prefetch_b8
    }
    // stage A: 128x32 f32 -> f16, 16 elements per thread, coalesced
#pragma unroll
    for (int i = 0; i < 16; ++i) {
      int idx = i * 256 + tid;
      int r = idx >> 5, c = idx & 31;
      int gm = blockM + r, gk = k0 + c;
      float v = 0.f;
      if (gm < M && gk < K) v = A[(size_t)gm * K + gk];
      As[r][c] = (_Float16)v;
    }
#if HAS_TDM
    if (fullK) {
      // stage B via Tensor Data Mover: 2D tile [32 rows x 64 cols], 2B elems
      if (tid == 0) {
        unsigned long long ga =
            (unsigned long long)(size_t)(B + (size_t)k0 * N + blockN);
        unsigned int lds = (unsigned int)(size_t)(&Bs[0][0]);
        uint32x4 g0;
        g0[0] = 1u;                               // count=1 valid user D#
        g0[1] = lds;                              // lds_addr (bytes)
        g0[2] = (unsigned int)ga;                 // global_addr[31:0]
        g0[3] = (unsigned int)(ga >> 32) | 0x80000000u;  // addr[56:32]|type=2
        int32x8 g1;
        g1[0] = (1 << 16)      // data_size = 2 bytes
              | (1 << 20)      // pad_enable
              | (4 << 22)      // pad_interval: 32 DWORDs (=128B = one row)
              | (3 << 25);     // pad_amount: 4 DWORDs (=16B -> 72-half stride)
        g1[1] = (N & 0xFFFF) << 16;                       // tensor_dim0 lo16
        g1[2] = (N >> 16) | (((K - k0) & 0xFFFF) << 16);  // dim0 hi | dim1 lo
        g1[3] = (((K - k0) >> 16) & 0xFFFF) | (64 << 16); // dim1 hi | tile_dim0
        g1[4] = 32;                                       // tile_dim1
        g1[5] = N;                                        // dim0_stride lo32
        g1[6] = 0;
        g1[7] = 0;
        int32x4 z4 = {};
#if __clang_major__ >= 23
        int32x8 z8 = {};
        __builtin_amdgcn_tensor_load_to_lds(g0, g1, z4, z4, z8, 0);
#else
        __builtin_amdgcn_tensor_load_to_lds(g0, g1, z4, z4, 0);
#endif
        __builtin_amdgcn_s_wait_tensorcnt(0);
      }
    } else
#endif
    {
      // fallback staging (also used for the K=16 first SAGE layer)
#pragma unroll
      for (int i = 0; i < 8; ++i) {
        int idx = i * 256 + tid;
        int r = idx >> 6, c = idx & 63;
        int gk = k0 + r;
        _Float16 v = (_Float16)0.f;
        if (gk < K) v = B[(size_t)gk * N + (blockN + c)];
        Bs[r][c] = v;
      }
    }
    __syncthreads();

    half16 af[2], bf[2];
#pragma unroll
    for (int t = 0; t < 2; ++t) {
#pragma unroll
      for (int e = 0; e < 16; ++e) {
        int kk = wk_idx(e, g);
        af[t][e] = As[wm * 32 + t * 16 + mr][kk];
        bf[t][e] = Bs[kk][wn * 32 + t * 16 + mr];
      }
    }
#pragma unroll
    for (int mt = 0; mt < 2; ++mt)
#pragma unroll
      for (int nt = 0; nt < 2; ++nt)
        acc[mt][nt] = __builtin_amdgcn_wmma_f32_16x16x32_f16(
            false, af[mt], false, bf[nt], (short)0, acc[mt][nt], false, false);
    __syncthreads();
  }

  // epilogue: C fragment m = r + 8*g, n = lane%16
#pragma unroll
  for (int mt = 0; mt < 2; ++mt)
#pragma unroll
    for (int nt = 0; nt < 2; ++nt)
#pragma unroll
      for (int r = 0; r < 8; ++r) {
        int m = blockM + wm * 32 + mt * 16 + r + 8 * g;
        int n = blockN + wn * 32 + nt * 16 + mr;
        if (m < M) {
          float v = acc[mt][nt][r];
          if constexpr (BIAS) v += bias[n];
          if constexpr (ADDC) v += Cadd[(size_t)m * N + n];
          if constexpr (RELU) v = fmaxf(v, 0.f);
          C[(size_t)m * N + n] = (OutT)v;
        }
      }
}

// ---------------------------------------------------------------------------
// Flash attention over packed sequences.  One wave per (query-tile, graph,
// head).  dh = 32 == one f16 WMMA K-step; keys processed 32 at a time with
// online softmax; P re-laid C-frag -> A-frag through LDS.  Fully unrolled so
// softmax state stays in flat registers (no v_movrel).
// q/k/v layout: row = l*N_GRAPHS + g, col = head*32 + d   (f16)
// ---------------------------------------------------------------------------
__launch_bounds__(32)
__global__ void attn_kernel(const _Float16* __restrict__ q,
                            const _Float16* __restrict__ k,
                            const _Float16* __restrict__ v,
                            const int* __restrict__ batches,
                            float* __restrict__ out) {
  __shared__ _Float16 p_lds[16][40];

  const int qt  = blockIdx.x;      // 0..19 query tile
  const int gph = blockIdx.y;      // 0..63 graph
  const int hd  = blockIdx.z;      // 0..7  head
  const int lane = threadIdx.x;
  const int g  = lane >> 4;
  const int mr = lane & 15;
  const int len = batches[gph + 1] - batches[gph];
  const float scale = 0.1767766952966369f;  // 1/sqrt(32)

  // Q fragment (A): row m = lane%16 -> query l = qt*16+m
  half16 aq;
  {
    int l = qt * 16 + mr;
    size_t base = ((size_t)l * N_GRAPHS + gph) * EMB + hd * DH;
#pragma unroll
    for (int e = 0; e < 16; ++e) aq[e] = q[base + wk_idx(e, g)];
  }

  float8 o0 = {}, o1 = {};
  float mrow[8], lrow[8];
#pragma unroll
  for (int r = 0; r < 8; ++r) { mrow[r] = -1e30f; lrow[r] = 0.f; }

  for (int kt = 0; kt < LMAX / 32; ++kt) {
    int j0 = kt * 32;

    // K^T fragments (B): lane column n = key-in-tile
    size_t b0 = ((size_t)(j0 + mr)      * N_GRAPHS + gph) * EMB + hd * DH;
    size_t b1 = ((size_t)(j0 + 16 + mr) * N_GRAPHS + gph) * EMB + hd * DH;
    if (kt + 1 < LMAX / 32) {   // prefetch next K/V tiles
      __builtin_prefetch(&k[b0 + (size_t)32 * N_GRAPHS * EMB], 0, 0);
      __builtin_prefetch(&v[b0 + (size_t)32 * N_GRAPHS * EMB], 0, 0);
    }
    half16 bk0, bk1;
#pragma unroll
    for (int e = 0; e < 16; ++e) {
      int kk = wk_idx(e, g);
      bk0[e] = k[b0 + kk];
      bk1[e] = k[b1 + kk];
    }
    float8 zc = {};
    float8 s0 = __builtin_amdgcn_wmma_f32_16x16x32_f16(false, aq, false, bk0,
                                                       (short)0, zc, false, false);
    float8 s1 = __builtin_amdgcn_wmma_f32_16x16x32_f16(false, aq, false, bk1,
                                                       (short)0, zc, false, false);

    const bool mask0 = (j0 + mr)      >= len;   // key padding mask (column-wise)
    const bool mask1 = (j0 + 16 + mr) >= len;

#pragma unroll
    for (int r = 0; r < 8; ++r) {
      float a = mask0 ? -1e9f : s0[r] * scale;
      float b = mask1 ? -1e9f : s1[r] * scale;
      // row max across the 16 lanes holding this score row
      float mt = fmaxf(a, b);
#pragma unroll
      for (int off = 1; off < 16; off <<= 1)
        mt = fmaxf(mt, __shfl_xor(mt, off, 32));
      float mnew  = fmaxf(mrow[r], mt);
      float alpha = __expf(mrow[r] - mnew);
      o0[r] *= alpha;
      o1[r] *= alpha;
      float p0 = __expf(a - mnew);
      float p1 = __expf(b - mnew);
      float ps = p0 + p1;
#pragma unroll
      for (int off = 1; off < 16; off <<= 1)
        ps += __shfl_xor(ps, off, 32);
      lrow[r] = lrow[r] * alpha + ps;
      mrow[r] = mnew;
      // C-frag -> LDS (row m = r+8g, key column)
      p_lds[r + 8 * g][mr]      = (_Float16)p0;
      p_lds[r + 8 * g][16 + mr] = (_Float16)p1;
    }
    __syncthreads();

    // P A-fragment from LDS; V B-fragments from global
    half16 ap, bv0, bv1;
#pragma unroll
    for (int e = 0; e < 16; ++e) {
      int kk = wk_idx(e, g);
      ap[e] = p_lds[mr][kk];
      size_t vb = ((size_t)(j0 + kk) * N_GRAPHS + gph) * EMB + hd * DH;
      bv0[e] = v[vb + mr];
      bv1[e] = v[vb + 16 + mr];
    }
    o0 = __builtin_amdgcn_wmma_f32_16x16x32_f16(false, ap, false, bv0,
                                                (short)0, o0, false, false);
    o1 = __builtin_amdgcn_wmma_f32_16x16x32_f16(false, ap, false, bv1,
                                                (short)0, o1, false, false);
    __syncthreads();
  }

#pragma unroll
  for (int r = 0; r < 8; ++r) {
    int m = r + 8 * g;
    int l = qt * 16 + m;
    float inv = 1.f / lrow[r];
    size_t base = ((size_t)l * N_GRAPHS + gph) * EMB + hd * DH;
    out[base + mr]      = o0[r] * inv;
    out[base + 16 + mr] = o1[r] * inv;
  }
}

// ---------------------------------------------------------------------------
// Small helper kernels
// ---------------------------------------------------------------------------
__global__ void zero_f32_kernel(float* p, long long n) {
  long long i = (long long)blockIdx.x * blockDim.x + threadIdx.x;
  if (i < n) p[i] = 0.f;
}

__global__ void f32_to_f16_kernel(const float* __restrict__ s,
                                  _Float16* __restrict__ d, long long n) {
  long long i = (long long)blockIdx.x * blockDim.x + threadIdx.x;
  if (i < n) d[i] = (_Float16)s[i];
}

__global__ void degree_kernel(const int* __restrict__ dst, float* deg, int ne) {
  int i = blockIdx.x * blockDim.x + threadIdx.x;
  if (i < ne) atomicAdd(&deg[dst[i]], 1.0f);
}

__global__ void scatter_kernel(const float* __restrict__ h,
                               const int* __restrict__ src,
                               const int* __restrict__ dst,
                               float* __restrict__ agg, int ne, int D) {
  long long i = (long long)blockIdx.x * blockDim.x + threadIdx.x;
  long long total = (long long)ne * D;
  if (i < total) {
    int e = (int)(i / D);
    int d = (int)(i % D);
    atomicAdd(&agg[(size_t)dst[e] * D + d], h[(size_t)src[e] * D + d]);
  }
}

__global__ void scale_agg_kernel(float* __restrict__ agg,
                                 const float* __restrict__ deg, int n, int D) {
  long long i = (long long)blockIdx.x * blockDim.x + threadIdx.x;
  long long total = (long long)n * D;
  if (i < total) {
    int row = (int)(i / D);
    agg[i] *= 1.f / fmaxf(deg[row], 1.f);
  }
}

__global__ void pack_kernel(const float* __restrict__ z,
                            const int* __restrict__ batches,
                            float* __restrict__ hpk) {
  int node = blockIdx.x;
  int d    = threadIdx.x;
  // binary search: largest g with batches[g] <= node
  int lo = 0, hi = N_GRAPHS;
  while (hi - lo > 1) {
    int mid = (lo + hi) >> 1;
    if (batches[mid] <= node) lo = mid; else hi = mid;
  }
  int pos = node - batches[lo];
  hpk[((size_t)pos * N_GRAPHS + lo) * EMB + d] = z[(size_t)node * EMB + d];
}

// block per row, 256 threads == EMB
__global__ void ln_kernel(const float* __restrict__ x,
                          const float* __restrict__ gam,
                          const float* __restrict__ bet,
                          float* __restrict__ out) {
  __shared__ float red[256];
  const int row = blockIdx.x;
  const int t   = threadIdx.x;
  float v = x[(size_t)row * EMB + t];
  red[t] = v;
  __syncthreads();
  for (int s = 128; s > 0; s >>= 1) {
    if (t < s) red[t] += red[t + s];
    __syncthreads();
  }
  float mean = red[0] * (1.f / EMB);
  __syncthreads();
  float d = v - mean;
  red[t] = d * d;
  __syncthreads();
  for (int s = 128; s > 0; s >>= 1) {
    if (t < s) red[t] += red[t + s];
    __syncthreads();
  }
  float var = red[0] * (1.f / EMB);
  out[(size_t)row * EMB + t] = d * rsqrtf(var + 1e-5f) * gam[t] + bet[t];
}

// ---------------------------------------------------------------------------
// Host-side orchestration
// ---------------------------------------------------------------------------
static inline size_t alignup(size_t x) { return (x + 255) & ~(size_t)255; }

extern "C" void kernel_launch(void* const* d_in, const int* in_sizes, int n_in,
                              void* d_out, int out_size, void* d_ws, size_t ws_size,
                              hipStream_t stream) {
  (void)in_sizes; (void)n_in; (void)out_size; (void)ws_size;

  const float* x        = (const float*)d_in[0];
  const int*   eidx     = (const int*)d_in[1];
  const int*   src      = eidx;
  const int*   dst      = eidx + N_EDGES;
  const int*   batches  = (const int*)d_in[2];
  // d_in[3] = L_max scalar (compile-time constant here)
  const float* sW0  = (const float*)d_in[4];
  const float* sWn0 = (const float*)d_in[5];
  const float* sb0  = (const float*)d_in[6];
  const float* sW1  = (const float*)d_in[7];
  const float* sWn1 = (const float*)d_in[8];
  const float* sb1  = (const float*)d_in[9];
  const float* sW2  = (const float*)d_in[10];
  const float* sWn2 = (const float*)d_in[11];
  const float* sb2  = (const float*)d_in[12];
  const float* Wq = (const float*)d_in[13];
  const float* Wk = (const float*)d_in[14];
  const float* Wv = (const float*)d_in[15];
  const float* bq = (const float*)d_in[16];
  const float* bk = (const float*)d_in[17];
  const float* bv = (const float*)d_in[18];
  const float* Wo = (const float*)d_in[19];
  const float* bo = (const float*)d_in[20];
  const float* ln1g = (const float*)d_in[21];
  const float* ln1b = (const float*)d_in[22];
  const float* ln2g = (const float*)d_in[23];
  const float* ln2b = (const float*)d_in[24];
  const float* W1 = (const float*)d_in[25];
  const float* b1 = (const float*)d_in[26];
  const float* W2 = (const float*)d_in[27];
  const float* b2 = (const float*)d_in[28];

  // ---- workspace carve-out ----
  char* cur = (char*)d_ws;
  auto take = [&](size_t bytes) { char* p = cur; cur += alignup(bytes); return p; };

  float* deg  = (float*)take((size_t)N_NODES * 4);
  float* agg  = (float*)take((size_t)N_NODES * EMB * 4);
  float* z0   = (float*)take((size_t)N_NODES * EMB * 4);
  float* z1   = (float*)take((size_t)N_NODES * EMB * 4);
  float* hbuf = (float*)take((size_t)LG * EMB * 4);
  float* tmp  = (float*)take((size_t)LG * EMB * 4);
  float* attn = (float*)take((size_t)LG * EMB * 4);
  float* mid  = (float*)take((size_t)LG * FFN_DIM * 4);
  _Float16* qh = (_Float16*)take((size_t)LG * EMB * 2);
  _Float16* kh = (_Float16*)take((size_t)LG * EMB * 2);
  _Float16* vh = (_Float16*)take((size_t)LG * EMB * 2);
  _Float16* sW0h  = (_Float16*)take((size_t)COLORS * EMB * 2);
  _Float16* sWn0h = (_Float16*)take((size_t)COLORS * EMB * 2);
  _Float16* sW1h  = (_Float16*)take((size_t)EMB * EMB * 2);
  _Float16* sWn1h = (_Float16*)take((size_t)EMB * EMB * 2);
  _Float16* sW2h  = (_Float16*)take((size_t)EMB * EMB * 2);
  _Float16* sWn2h = (_Float16*)take((size_t)EMB * EMB * 2);
  _Float16* Wqh = (_Float16*)take((size_t)LAYERS * EMB * EMB * 2);
  _Float16* Wkh = (_Float16*)take((size_t)LAYERS * EMB * EMB * 2);
  _Float16* Wvh = (_Float16*)take((size_t)LAYERS * EMB * EMB * 2);
  _Float16* Woh = (_Float16*)take((size_t)LAYERS * EMB * EMB * 2);
  _Float16* W1h = (_Float16*)take((size_t)LAYERS * EMB * FFN_DIM * 2);
  _Float16* W2h = (_Float16*)take((size_t)LAYERS * FFN_DIM * EMB * 2);

  auto cvt = [&](const float* s, _Float16* d, long long n) {
    f32_to_f16_kernel<<<(unsigned)((n + 255) / 256), 256, 0, stream>>>(s, d, n);
  };
  cvt(sW0, sW0h, COLORS * EMB);   cvt(sWn0, sWn0h, COLORS * EMB);
  cvt(sW1, sW1h, EMB * EMB);      cvt(sWn1, sWn1h, EMB * EMB);
  cvt(sW2, sW2h, EMB * EMB);      cvt(sWn2, sWn2h, EMB * EMB);
  cvt(Wq, Wqh, (long long)LAYERS * EMB * EMB);
  cvt(Wk, Wkh, (long long)LAYERS * EMB * EMB);
  cvt(Wv, Wvh, (long long)LAYERS * EMB * EMB);
  cvt(Wo, Woh, (long long)LAYERS * EMB * EMB);
  cvt(W1, W1h, (long long)LAYERS * EMB * FFN_DIM);
  cvt(W2, W2h, (long long)LAYERS * FFN_DIM * EMB);

  auto zero = [&](float* p, long long n) {
    zero_f32_kernel<<<(unsigned)((n + 255) / 256), 256, 0, stream>>>(p, n);
  };

  // ---- degrees ----
  zero(deg, N_NODES);
  degree_kernel<<<(N_EDGES + 255) / 256, 256, 0, stream>>>(dst, deg, N_EDGES);

  dim3 gemmBlk(256);
  auto gemm_grid = [](int M, int N) { return dim3(M / 128, N / 64); };

  // ---- SAGE layer 0 (K = 16) ----
  zero(agg, (long long)N_NODES * COLORS);
  scatter_kernel<<<(unsigned)(((long long)N_EDGES * COLORS + 255) / 256), 256, 0, stream>>>(
      x, src, dst, agg, N_EDGES, COLORS);
  scale_agg_kernel<<<(unsigned)(((long long)N_NODES * COLORS + 255) / 256), 256, 0, stream>>>(
      agg, deg, N_NODES, COLORS);
  gemm_wmma_kernel<false, false, true, float><<<gemm_grid(N_NODES, EMB), gemmBlk, 0, stream>>>(
      agg, sWn0h, sb0, nullptr, z0, N_NODES, EMB, COLORS);
  gemm_wmma_kernel<true, true, false, float><<<gemm_grid(N_NODES, EMB), gemmBlk, 0, stream>>>(
      x, sW0h, nullptr, z0, z0, N_NODES, EMB, COLORS);

  // ---- SAGE layer 1 ----
  zero(agg, (long long)N_NODES * EMB);
  scatter_kernel<<<(unsigned)(((long long)N_EDGES * EMB + 255) / 256), 256, 0, stream>>>(
      z0, src, dst, agg, N_EDGES, EMB);
  scale_agg_kernel<<<(unsigned)(((long long)N_NODES * EMB + 255) / 256), 256, 0, stream>>>(
      agg, deg, N_NODES, EMB);
  gemm_wmma_kernel<false, false, true, float><<<gemm_grid(N_NODES, EMB), gemmBlk, 0, stream>>>(
      agg, sWn1h, sb1, nullptr, z1, N_NODES, EMB, EMB);
  gemm_wmma_kernel<true, true, false, float><<<gemm_grid(N_NODES, EMB), gemmBlk, 0, stream>>>(
      z0, sW1h, nullptr, z1, z1, N_NODES, EMB, EMB);

  // ---- SAGE layer 2 (no activation) ----
  zero(agg, (long long)N_NODES * EMB);
  scatter_kernel<<<(unsigned)(((long long)N_EDGES * EMB + 255) / 256), 256, 0, stream>>>(
      z1, src, dst, agg, N_EDGES, EMB);
  scale_agg_kernel<<<(unsigned)(((long long)N_NODES * EMB + 255) / 256), 256, 0, stream>>>(
      agg, deg, N_NODES, EMB);
  gemm_wmma_kernel<false, false, true, float><<<gemm_grid(N_NODES, EMB), gemmBlk, 0, stream>>>(
      agg, sWn2h, sb2, nullptr, z0, N_NODES, EMB, EMB);
  gemm_wmma_kernel<false, true, false, float><<<gemm_grid(N_NODES, EMB), gemmBlk, 0, stream>>>(
      z1, sW2h, nullptr, z0, z0, N_NODES, EMB, EMB);

  // ---- pack to [LMAX, N_GRAPHS, EMB] (seq-first, zero padded) ----
  zero(hbuf, (long long)LG * EMB);
  pack_kernel<<<N_NODES, EMB, 0, stream>>>(z0, batches, hbuf);

  // ---- transformer layers ----
  for (int l = 0; l < LAYERS; ++l) {
    const size_t wo = (size_t)l * EMB * EMB;
    // Q/K/V projections, f16 output for WMMA attention
    gemm_wmma_kernel<false, false, true, _Float16><<<gemm_grid(LG, EMB), gemmBlk, 0, stream>>>(
        hbuf, Wqh + wo, bq + (size_t)l * EMB, nullptr, qh, LG, EMB, EMB);
    gemm_wmma_kernel<false, false, true, _Float16><<<gemm_grid(LG, EMB), gemmBlk, 0, stream>>>(
        hbuf, Wkh + wo, bk + (size_t)l * EMB, nullptr, kh, LG, EMB, EMB);
    gemm_wmma_kernel<false, false, true, _Float16><<<gemm_grid(LG, EMB), gemmBlk, 0, stream>>>(
        hbuf, Wvh + wo, bv + (size_t)l * EMB, nullptr, vh, LG, EMB, EMB);

    // flash attention: (query tile, graph, head)
    attn_kernel<<<dim3(LMAX / 16, N_GRAPHS, HEADS), 32, 0, stream>>>(
        qh, kh, vh, batches, attn);

    // output projection + residual, then LN1
    gemm_wmma_kernel<false, true, true, float><<<gemm_grid(LG, EMB), gemmBlk, 0, stream>>>(
        attn, Woh + wo, bo + (size_t)l * EMB, hbuf, tmp, LG, EMB, EMB);
    ln_kernel<<<LG, EMB, 0, stream>>>(tmp, ln1g + (size_t)l * EMB, ln1b + (size_t)l * EMB, hbuf);

    // FFN: relu(h@W1+b1)@W2 + b2 + residual, then LN2
    gemm_wmma_kernel<true, false, true, float><<<gemm_grid(LG, FFN_DIM), gemmBlk, 0, stream>>>(
        hbuf, W1h + (size_t)l * EMB * FFN_DIM, b1 + (size_t)l * FFN_DIM, nullptr,
        mid, LG, FFN_DIM, EMB);
    gemm_wmma_kernel<false, true, true, float><<<gemm_grid(LG, EMB), gemmBlk, 0, stream>>>(
        mid, W2h + (size_t)l * FFN_DIM * EMB, b2 + (size_t)l * EMB, hbuf,
        tmp, LG, EMB, FFN_DIM);
    float* lnout = (l == LAYERS - 1) ? (float*)d_out : hbuf;
    ln_kernel<<<LG, EMB, 0, stream>>>(tmp, ln2g + (size_t)l * EMB, ln2b + (size_t)l * EMB, lnout);
  }
}
